// MultiheadAttention_86011015069974
// MI455X (gfx1250) — matrix operand order repaired and checked by reading, not compiled, not run on hardware
//
#include <hip/hip_runtime.h>

#define S_LEN 2048
#define DM    2048
#define NH    16
#define HD    128

typedef __attribute__((ext_vector_type(16))) __bf16         v16bf;
typedef __attribute__((ext_vector_type(8)))  float          f32x8;
typedef __attribute__((ext_vector_type(4)))  unsigned int   u32x4;
typedef __attribute__((ext_vector_type(4)))  float          f32x4;
typedef __attribute__((ext_vector_type(4)))  unsigned short u16x4;
typedef int v4i __attribute__((vector_size(16)));  // async builtin's pointee type

#if __has_builtin(__builtin_amdgcn_global_load_async_to_lds_b128) && \
    __has_builtin(__builtin_amdgcn_s_wait_asynccnt)
#define HAVE_ASYNC 1
#else
#define HAVE_ASYNC 0
#endif

union Frag { v16bf v; u32x4 u[2]; };

__device__ inline f32x8 wmma_bf16(const Frag& a, const Frag& b, f32x8 c) {
  return __builtin_amdgcn_wmma_f32_16x16x32_bf16(false, a.v, false, b.v,
                                                 (short)0, c, false, false);
}

__device__ inline unsigned short f2bf(float f) {
  unsigned int u = __builtin_bit_cast(unsigned int, f);
  u += 0x7FFFu + ((u >> 16) & 1u);  // round to nearest even
  return (unsigned short)(u >> 16);
}

__device__ inline f32x8 vzero() {
  f32x8 v;
#pragma unroll
  for (int i = 0; i < 8; i++) v[i] = 0.0f;
  return v;
}

// 16-byte global -> LDS copy: async DMA (ASYNCcnt) when available
__device__ inline void copy16(const unsigned short* g, unsigned short* l) {
#if HAVE_ASYNC
  __builtin_amdgcn_global_load_async_to_lds_b128(
      (__attribute__((address_space(1))) v4i*)(void*)g,
      (__attribute__((address_space(3))) v4i*)(void*)l, 0, 0);
#else
  *(u32x4*)l = *(const u32x4*)g;
#endif
}

__device__ inline void tile_wait() {
#if HAVE_ASYNC
  __builtin_amdgcn_s_wait_asynccnt(0);
#endif
  __syncthreads();
}

// row-of-16 max reduction via DPP row_xmask (pure VALU, no LDS)
template <int CTRL>
__device__ inline float dpp_fmax(float x) {
  int y = __builtin_amdgcn_update_dpp(0, __builtin_bit_cast(int, x),
                                      CTRL, 0xf, 0xf, true);
  return fmaxf(x, __builtin_bit_cast(float, y));
}
__device__ inline float rowmax16(float x) {
  x = dpp_fmax<0x161>(x);  // row_xmask:1
  x = dpp_fmax<0x162>(x);  // row_xmask:2
  x = dpp_fmax<0x164>(x);  // row_xmask:4
  x = dpp_fmax<0x168>(x);  // row_xmask:8
  return x;
}

// ---------------- lengths from key_padding_mask ----------------
__global__ __launch_bounds__(256) void compute_len(const int* __restrict__ kpm,
                                                   int* __restrict__ lens) {
  __shared__ int red[256];
  for (int b = 0; b < 2; b++) {
    int s = 0;
    for (int i = threadIdx.x; i < S_LEN; i += 256) s += (kpm[b * S_LEN + i] != 0);
    red[threadIdx.x] = s;
    __syncthreads();
    for (int o = 128; o > 0; o >>= 1) {
      if ((int)threadIdx.x < o) red[threadIdx.x] += red[threadIdx.x + o];
      __syncthreads();
    }
    if (threadIdx.x == 0) lens[b] = red[0];
    __syncthreads();
  }
}

// ---------------- fp32 -> bf16 conversion ----------------
__global__ __launch_bounds__(256) void convert_kernel(const float* __restrict__ src,
                                                      unsigned short* __restrict__ dst,
                                                      int n) {
  int i = (blockIdx.x * 256 + threadIdx.x) * 4;
  if (i < n) {
    f32x4 f = *(const f32x4*)(src + i);
    u16x4 o;
#pragma unroll
    for (int j = 0; j < 4; j++) o[j] = f2bf(f[j]);
    *(u16x4*)(dst + i) = o;
  }
}

// ---------------- GEMM tile loader (128x32 A, 256x32 B per K-step) ----------------
__device__ inline void issue_gemm_tiles(const unsigned short* A, const unsigned short* Bm,
                                        unsigned short* as, unsigned short* bs,
                                        int K, int m0, int n0, int ko, int lr, int lc) {
  const unsigned short* pa = A + (size_t)(m0 + lr) * K + ko + lc;
  copy16(pa, as + lr * 40 + lc);
  copy16(pa + 8, as + lr * 40 + lc + 8);
  const unsigned short* pb0 = Bm + (size_t)(n0 + lr) * K + ko + lc;
  copy16(pb0, bs + lr * 40 + lc);
  copy16(pb0 + 8, bs + lr * 40 + lc + 8);
  const unsigned short* pb1 = Bm + (size_t)(n0 + lr + 128) * K + ko + lc;
  copy16(pb1, bs + (lr + 128) * 40 + lc);
  copy16(pb1 + 8, bs + (lr + 128) * 40 + lc + 8);
}

// ---------------- tiled bf16 WMMA GEMM: C[m,n] = sum_k A[m,k]*B[n,k] ----------------
// block tile 128x256, wave tile 64x64, double-buffered async LDS pipeline
// MODE 0: QKV projection epilogue; MODE 1: output projection epilogue (fp32)
template <int MODE>
__global__ __launch_bounds__(256) void gemm_kernel(
    const unsigned short* __restrict__ A,   // [M][K] bf16
    const unsigned short* __restrict__ Bm,  // [N][K] bf16
    const float* __restrict__ bias,         // [N]
    float* __restrict__ Cout,
    unsigned short* __restrict__ q_ws,
    unsigned short* __restrict__ k_ws,
    unsigned short* __restrict__ vt_ws,
    const int* __restrict__ lengths,
    int K) {
  __shared__ unsigned short As[2][128 * 40];
  __shared__ unsigned short Bs[2][256 * 40];

  const int tid  = threadIdx.x;
  const int lane = tid & 31, wave = tid >> 5;
  const int l16  = lane & 15, half = lane >> 4;
  const int wm   = wave >> 2, wn = wave & 3;  // 2x4 waves of 64x64
  const int m0   = blockIdx.y * 128, n0 = blockIdx.x * 256;

  const int lr = tid >> 1;
  const int lc = (tid & 1) * 16;

  f32x8 acc[4][4];
#pragma unroll
  for (int i = 0; i < 4; i++)
#pragma unroll
    for (int j = 0; j < 4; j++) acc[i][j] = vzero();

  const int nk = K >> 5;
  issue_gemm_tiles(A, Bm, As[0], Bs[0], K, m0, n0, 0, lr, lc);

  for (int kt = 0; kt < nk; ++kt) {
    const int cur = kt & 1;
    tile_wait();  // tile kt resident in buffer `cur`
    if (kt + 1 < nk)
      issue_gemm_tiles(A, Bm, As[cur ^ 1], Bs[cur ^ 1], K, m0, n0,
                       (kt + 1) * 32, lr, lc);

    const unsigned short* as = As[cur];
    const unsigned short* bs = Bs[cur];
    Frag af[4];
#pragma unroll
    for (int mi = 0; mi < 4; mi++) {
      int base = (wm * 64 + mi * 16 + l16) * 40;
      af[mi].u[0] = *(const u32x4*)&as[base + half * 16];
      af[mi].u[1] = *(const u32x4*)&as[base + half * 16 + 8];
    }
#pragma unroll
    for (int ni = 0; ni < 4; ni++) {
      Frag bfr;
      int bb = (wn * 64 + ni * 16 + l16) * 40 + half * 16;
      bfr.u[0] = *(const u32x4*)&bs[bb];
      bfr.u[1] = *(const u32x4*)&bs[bb + 8];
#pragma unroll
      for (int mi = 0; mi < 4; mi++)
        acc[mi][ni] = wmma_bf16(af[mi], bfr, acc[mi][ni]);
    }
  }

  // epilogue: row = acc idx + 8*half, col = lane&15 (ISA C-layout)
#pragma unroll
  for (int mi = 0; mi < 4; mi++) {
#pragma unroll
    for (int ni = 0; ni < 4; ni++) {
      const int cg0 = n0 + wn * 64 + ni * 16;                 // tile col base
      const int rg0 = m0 + wm * 64 + mi * 16 + 8 * half;      // this half's row base
      const float bv = bias[cg0 + l16];
      if (MODE == 0) {
        const int sec = cg0 >> 11;                 // 0=q 1=k 2=v (uniform per tile)
        const int c2  = (cg0 & 2047) + l16;
        const int head = c2 >> 7, hdc = c2 & 127;
        const int b = rg0 >> 11, si0 = rg0 & 2047;
        const int len = lengths[b];
        if (sec == 2) {
          // v transposed: 8 consecutive si -> one aligned 16B store
          u32x4 pk;
#pragma unroll
          for (int v = 0; v < 8; v += 2) {
            unsigned int lo = (si0 + v < len) ? f2bf(acc[mi][ni][v] + bv) : 0u;
            unsigned int hi = (si0 + v + 1 < len) ? f2bf(acc[mi][ni][v + 1] + bv) : 0u;
            pk[v >> 1] = lo | (hi << 16);
          }
          *(u32x4*)&vt_ws[((size_t)((b * NH + head) * HD + hdc)) * S_LEN + si0] = pk;
        } else {
          const size_t base = ((size_t)((b * NH + head) * S_LEN + si0)) * HD + hdc;
          if (sec == 0) {
#pragma unroll
            for (int v = 0; v < 8; v++)
              q_ws[base + (size_t)v * HD] = f2bf(acc[mi][ni][v] + bv);
          } else {
#pragma unroll
            for (int v = 0; v < 8; v++)
              k_ws[base + (size_t)v * HD] =
                  (si0 + v < len) ? f2bf(acc[mi][ni][v] + bv) : (unsigned short)0;
          }
        }
      } else {
        float* cp = Cout + (size_t)rg0 * DM + cg0 + l16;
#pragma unroll
        for (int v = 0; v < 8; v++) cp[(size_t)v * DM] = acc[mi][ni][v] + bv;
      }
    }
  }
}

// ---------------- attention K/V tile loader (32 keys) ----------------
__device__ inline void issue_kv(const unsigned short* kbase, const unsigned short* vbase,
                                unsigned short* Klb, unsigned short* Vlb,
                                int kt, int tid) {
  const int krow = tid >> 2, kcol = (tid & 3) * 32;
  const unsigned short* pk = kbase + (size_t)(kt * 32 + krow) * HD + kcol;
  unsigned short* lk = Klb + krow * 136 + kcol;
  copy16(pk, lk);
  copy16(pk + 8, lk + 8);
  copy16(pk + 16, lk + 16);
  copy16(pk + 24, lk + 24);
  const unsigned short* pv = vbase + (size_t)tid * S_LEN + kt * 32;
  unsigned short* lv = Vlb + tid * 40;
  copy16(pv, lv);
  copy16(pv + 8, lv + 8);
  copy16(pv + 16, lv + 16);
  copy16(pv + 24, lv + 24);
}

// ---------------- flash attention with ALiBi + causal + key padding ----------------
__global__ __launch_bounds__(128) void attn_kernel(
    const unsigned short* __restrict__ q_ws,   // [b*h][s][hd]
    const unsigned short* __restrict__ k_ws,   // [b*h][s][hd]
    const unsigned short* __restrict__ vt_ws,  // [b*h][hd][s]
    unsigned short* __restrict__ attn_ws,      // [b][s][d] bf16
    const int* __restrict__ lengths) {
  __shared__ unsigned short Kl[2][32 * 136];   // 32 keys x 128 hd
  __shared__ unsigned short Vl[2][128 * 40];   // 128 hd  x 32 keys
  __shared__ unsigned short Pb[4][16 * 32];    // per-wave P tile

  const int tid  = threadIdx.x;
  const int lane = tid & 31, wave = tid >> 5;
  const int l16  = lane & 15, half = lane >> 4;

  const int qt = blockIdx.x & 31;
  const int bh = blockIdx.x >> 5;
  const int h  = bh & 15, b = bh >> 4;
  const int q0 = qt * 64 + wave * 16;

  const int len = lengths[b];
  const float slope  = exp2f(-0.5f * (float)(h + 1));
  const float sscale = 0.08838834764831843f;  // 1/sqrt(128)

  const unsigned short* qbase = q_ws + ((size_t)bh * S_LEN) * HD;
  const unsigned short* kbase = k_ws + ((size_t)bh * S_LEN) * HD;
  const unsigned short* vbase = vt_ws + ((size_t)bh * HD) * S_LEN;

  Frag qf[4];
#pragma unroll
  for (int j = 0; j < 4; j++) {
    const unsigned short* p = qbase + (size_t)(q0 + l16) * HD + 32 * j + half * 16;
    qf[j].u[0] = *(const u32x4*)p;
    qf[j].u[1] = *(const u32x4*)(p + 8);
  }

  Frag ones;  // B fragment of 1.0 bf16 -> rowsum(P) via WMMA
  {
    u32x4 o1;
#pragma unroll
    for (int i = 0; i < 4; i++) o1[i] = 0x3F803F80u;
    ones.u[0] = o1;
    ones.u[1] = o1;
  }

  f32x8 o[8];
#pragma unroll
  for (int t = 0; t < 8; t++) o[t] = vzero();
  f32x8 ls = vzero();  // rowsum accumulator (all 16 cols identical)
  float m[8];
#pragma unroll
  for (int v = 0; v < 8; v++) m[v] = -1e30f;

  int kt_end = qt * 2 + 2;       // causal bound (32-key tiles)
  int kcap   = (len + 31) >> 5;  // padding bound
  if (kt_end > kcap) kt_end = kcap;

  issue_kv(kbase, vbase, Kl[0], Vl[0], 0, tid);

  for (int kt = 0; kt < kt_end; ++kt) {
    const int cur = kt & 1;
    tile_wait();  // tile kt resident
    if (kt + 1 < kt_end) issue_kv(kbase, vbase, Kl[cur ^ 1], Vl[cur ^ 1], kt + 1, tid);

    const unsigned short* kl = Kl[cur];
    const unsigned short* vl = Vl[cur];

    // S = Q K^T over hd=128: two 16x16 key tiles
    f32x8 s0 = vzero(), s1 = vzero();
#pragma unroll
    for (int j = 0; j < 4; j++) {
      int co = 32 * j + half * 16;
      Frag kb, kb2;
      kb.u[0]  = *(const u32x4*)&kl[l16 * 136 + co];
      kb.u[1]  = *(const u32x4*)&kl[l16 * 136 + co + 8];
      kb2.u[0] = *(const u32x4*)&kl[(16 + l16) * 136 + co];
      kb2.u[1] = *(const u32x4*)&kl[(16 + l16) * 136 + co + 8];
      s0 = wmma_bf16(qf[j], kb, s0);
      s1 = wmma_bf16(qf[j], kb2, s1);
    }

    const int col0 = kt * 32 + l16;
    const int col1 = col0 + 16;
#pragma unroll
    for (int v = 0; v < 8; v++) {
      int rw = q0 + v + 8 * half;
      float a = s0[v] * sscale + slope * (float)(col0 - (S_LEN - 1));
      float c = s1[v] * sscale + slope * (float)(col1 - (S_LEN - 1));
      if (col0 > rw || col0 >= len) a = -1e30f;
      if (col1 > rw || col1 >= len) c = -1e30f;
      float rm    = rowmax16(fmaxf(a, c));  // DPP row_xmask reduction
      float mn    = fmaxf(m[v], rm);
      float alpha = __expf(m[v] - mn);
      m[v] = mn;
      float p0 = __expf(a - mn);
      float p1 = __expf(c - mn);
      ls[v] *= alpha;
#pragma unroll
      for (int t = 0; t < 8; t++) o[t][v] *= alpha;
      Pb[wave][(v + 8 * half) * 32 + l16]      = f2bf(p0);
      Pb[wave][(v + 8 * half) * 32 + l16 + 16] = f2bf(p1);
    }

    Frag pf;
    pf.u[0] = *(const u32x4*)&Pb[wave][l16 * 32 + half * 16];
    pf.u[1] = *(const u32x4*)&Pb[wave][l16 * 32 + half * 16 + 8];

    ls = wmma_bf16(pf, ones, ls);  // l += rowsum(P), replicated per row
#pragma unroll
    for (int t = 0; t < 8; t++) {
      Frag vf;
      vf.u[0] = *(const u32x4*)&vl[(t * 16 + l16) * 40 + half * 16];
      vf.u[1] = *(const u32x4*)&vl[(t * 16 + l16) * 40 + half * 16 + 8];
      o[t] = wmma_bf16(pf, vf, o[t]);
    }
  }

  // epilogue: normalize and store bf16 at [b][s][h*hd + col]
#pragma unroll
  for (int v = 0; v < 8; v++) {
    float inv = 1.0f / ls[v];
    int rw = q0 + v + 8 * half;
    size_t rowbase = ((size_t)b * S_LEN + rw) * DM + h * HD;
#pragma unroll
    for (int t = 0; t < 8; t++)
      attn_ws[rowbase + t * 16 + l16] = f2bf(o[t][v] * inv);
  }
}

// ---------------- launch ----------------
extern "C" void kernel_launch(void* const* d_in, const int* in_sizes, int n_in,
                              void* d_out, int out_size, void* d_ws, size_t ws_size,
                              hipStream_t stream) {
  const float* x     = (const float*)d_in[0];
  const int*   kpm   = (const int*)d_in[1];
  const float* wqkv  = (const float*)d_in[2];
  const float* wqkvb = (const float*)d_in[3];
  const float* wo    = (const float*)d_in[4];
  const float* wob   = (const float*)d_in[5];

  char* ws = (char*)d_ws;
  int* lens = (int*)ws;
  unsigned short* xb    = (unsigned short*)(ws + 256);
  unsigned short* wqb   = xb + (size_t)4096 * 2048;     // x bf16
  unsigned short* wob16 = wqb + (size_t)6144 * 2048;    // Wqkv bf16
  unsigned short* qw    = wob16 + (size_t)2048 * 2048;  // out_w bf16
  unsigned short* kw    = qw + (size_t)2 * 16 * 2048 * 128;
  unsigned short* vtw   = kw + (size_t)2 * 16 * 2048 * 128;
  unsigned short* aw    = vtw + (size_t)2 * 16 * 2048 * 128;

  compute_len<<<1, 256, 0, stream>>>(kpm, lens);
  convert_kernel<<<(4096 * 2048 / 4 + 255) / 256, 256, 0, stream>>>(x, xb, 4096 * 2048);
  convert_kernel<<<(6144 * 2048 / 4 + 255) / 256, 256, 0, stream>>>(wqkv, wqb, 6144 * 2048);
  convert_kernel<<<(2048 * 2048 / 4 + 255) / 256, 256, 0, stream>>>(wo, wob16, 2048 * 2048);
  gemm_kernel<0><<<dim3(24, 32), 256, 0, stream>>>(xb, wqb, wqkvb, nullptr,
                                                   qw, kw, vtw, lens, 2048);
  attn_kernel<<<1024, 128, 0, stream>>>(qw, kw, vtw, aw, lens);
  gemm_kernel<1><<<dim3(8, 32), 256, 0, stream>>>(aw, wob16, wob, (float*)d_out,
                                                  nullptr, nullptr, nullptr, lens, 2048);
}
